// R2Former_63127429317049
// MI455X (gfx1250) — compile-verified, matrix operand order checked by reading, not run on hardware
//
#include <hip/hip_runtime.h>
#include <hip/hip_bf16.h>

// ---------------------------------------------------------------------------
// R2Former-style re-ranker, MI455X (gfx1250) port.
// All large GEMMs go through v_wmma_f32_16x16x32_bf16 (bf16 in, f32 accum).
// Everything fits in the 192MB L2 -> compute bound -> WMMA everywhere.
// ---------------------------------------------------------------------------

typedef unsigned short u16;
typedef __attribute__((ext_vector_type(16))) __bf16 v16bf;
typedef __attribute__((ext_vector_type(8)))  float  v8f;
typedef __attribute__((ext_vector_type(8)))  unsigned v8u;

static __device__ __forceinline__ u16 f2bf(float f) {
    unsigned u = __builtin_bit_cast(unsigned, f);
    return (u16)((u + 0x7FFFu + ((u >> 16) & 1u)) >> 16);   // RNE
}
static __device__ __forceinline__ float bf2f(u16 h) {
    unsigned u = ((unsigned)h) << 16;
    return __builtin_bit_cast(float, u);
}
static __device__ __forceinline__ float wred(float v) {
#pragma unroll
    for (int d = 1; d < 32; d <<= 1) v += __shfl_xor(v, d, 32);
    return v;
}
static __device__ __forceinline__ v8u load_frag16(const u16* p) {
    // p is 16B aligned; fragment = 8 bf16 at p[0..7] and 8 at p[16..23]
    uint4 a = *(const uint4*)p;
    uint4 b = *(const uint4*)(p + 16);
    v8u f;
    f[0]=a.x; f[1]=a.y; f[2]=a.z; f[3]=a.w;
    f[4]=b.x; f[5]=b.y; f[6]=b.z; f[7]=b.w;
    return f;
}
static __device__ __forceinline__ v8f wmma_bf16(v8u a, v8u b, v8f c) {
    return __builtin_amdgcn_wmma_f32_16x16x32_bf16(
        false, __builtin_bit_cast(v16bf, a),
        false, __builtin_bit_cast(v16bf, b),
        (short)0, c, false, false);
}
static __device__ __forceinline__ float clamp01(float v) {
    return fminf(fmaxf(v, 0.f), 1.f);
}
// 384-dim 2D sin/cos positional encoding: [sin_x(96),cos_x(96),sin_y(96),cos_y(96)]
static __device__ __forceinline__ float pe384(int j, float px, float py) {
    float pos = (j < 192) ? px : py;
    int jj = (j < 192) ? j : (j - 192);
    int d  = (jj < 96) ? jj : (jj - 96);
    float omega = __powf(10000.f, -(float)d * (1.f / 96.f));
    float a = pos * omega;
    return (jj < 96) ? __sinf(a) : __cosf(a);
}

// ---------------------------------------------------------------------------
// Generic bf16 WMMA GEMM:  C = act(A[MxK] @ WT[NxK]^T + bias) (+residual)
//   - 256 threads = 8 waves; workgroup tile 64(M) x 128(N); K step 32
//   - GELU: exact gelu;  RES: Cf += result in place;  OBF: write bf16 out
//   - grid.z batches with element strides (used by correlation GEMM)
// Requires: M%64==0, N%128==0, K%32==0 (all call sites padded accordingly)
// ---------------------------------------------------------------------------
template <int GELU, int RES, int OBF>
__global__ __launch_bounds__(256) void gemm_wmma_k(
    const u16* __restrict__ A, const u16* __restrict__ WT,
    const float* __restrict__ bias, float* __restrict__ Cf, u16* __restrict__ Cb,
    int M, int N, int Kd, long long aStride, long long wStride, long long cStride)
{
    __shared__ __align__(16) u16 As[64][40];
    __shared__ __align__(16) u16 Ws[128][40];
    const int bz = blockIdx.z;
    const u16* Ab = A  + (long long)bz * aStride;
    const u16* Wb = WT + (long long)bz * wStride;
    const long long cbase = (long long)bz * cStride;
    const int m0 = blockIdx.y * 64, n0 = blockIdx.x * 128;
    const int tid = threadIdx.x;
    const int wave = tid >> 5, lane = tid & 31;
    const int wm = wave >> 1, wn = wave & 1;
    const int lane15 = lane & 15, kb = (lane < 16) ? 0 : 8;

    v8f acc[4] = {};
    for (int k0 = 0; k0 < Kd; k0 += 32) {
        { // A tile: 64x32 bf16, one 16B chunk per thread
            int r = tid >> 2, cc = (tid & 3) * 8;
            *(uint4*)&As[r][cc] = *(const uint4*)(Ab + (long long)(m0 + r) * Kd + k0 + cc);
        }
#pragma unroll
        for (int i = 0; i < 2; i++) { // W tile (pre-transposed [N][K]): 128x32
            int c = tid + i * 256;
            int r = c >> 2, cc = (c & 3) * 8;
            *(uint4*)&Ws[r][cc] = *(const uint4*)(Wb + (long long)(n0 + r) * Kd + k0 + cc);
        }
        if (k0 + 32 < Kd)
            __builtin_prefetch(Ab + (long long)(m0 + (tid >> 2)) * Kd + k0 + 32, 0, 1);
        __syncthreads();
        v8u af = load_frag16(&As[wm * 16 + lane15][kb]);
#pragma unroll
        for (int nt = 0; nt < 4; nt++) {
            v8u bf_ = load_frag16(&Ws[wn * 64 + nt * 16 + lane15][kb]);
            acc[nt] = wmma_bf16(af, bf_, acc[nt]);
        }
        __syncthreads();
    }
    const int rb = m0 + wm * 16 + ((lane < 16) ? 0 : 8);
#pragma unroll
    for (int nt = 0; nt < 4; nt++) {
        int col = n0 + wn * 64 + nt * 16 + lane15;
        float bv = bias ? bias[col] : 0.f;
#pragma unroll
        for (int i = 0; i < 8; i++) {
            long long idx = cbase + (long long)(rb + i) * N + col;
            float v = acc[nt][i] + bv;
            if (GELU) v = 0.5f * v * (1.f + erff(v * 0.70710678118f));
            if (RES)  v += Cf[idx];
            if (OBF)  Cb[idx] = f2bf(v);
            else      Cf[idx] = v;
        }
    }
}

// ---------------------------------------------------------------------------
// Weight prep: f32 [K][N] -> bf16 [N][K] (transposed, WMMA-B friendly)
// ---------------------------------------------------------------------------
__global__ void prep_wt_k(const float* __restrict__ w, u16* __restrict__ wt, int K, int N)
{
    long long t = (long long)blockIdx.x * blockDim.x + threadIdx.x;
    if (t >= (long long)K * N) return;
    int k = (int)(t / N), n = (int)(t % N);
    wt[(long long)n * K + k] = f2bf(w[t]);
}

// ---------------------------------------------------------------------------
// Global cosine score (B=16, 256-dim)
// ---------------------------------------------------------------------------
__global__ __launch_bounds__(256) void gscore_k(const float* __restrict__ xg,
                                                const float* __restrict__ yg,
                                                float* __restrict__ gs)
{
    __shared__ float rxy[8], rxx[8], ryy[8];
    int b = blockIdx.x, t = threadIdx.x;
    float x = xg[b * 256 + t], y = yg[b * 256 + t];
    float sxy = wred(x * y), sxx = wred(x * x), syy = wred(y * y);
    if ((t & 31) == 0) { rxy[t >> 5] = sxy; rxx[t >> 5] = sxx; ryy[t >> 5] = syy; }
    __syncthreads();
    if (t == 0) {
        float a = 0, bb = 0, c = 0;
        for (int i = 0; i < 8; i++) { a += rxy[i]; bb += rxx[i]; c += ryy[i]; }
        gs[b] = a / (sqrtf(bb) * sqrtf(c));
    }
}

// ---------------------------------------------------------------------------
// Normalize descriptors (cols 3..130) -> bf16 [16][512][128] (rows >=500 zero)
// one wave per output row
// ---------------------------------------------------------------------------
__global__ __launch_bounds__(256) void norm_desc_k(const float* __restrict__ R,
                                                   u16* __restrict__ out, int Nn, int F)
{
    int row = blockIdx.x * 8 + (threadIdx.x >> 5);
    if (row >= 16 * 512) return;
    int lane = threadIdx.x & 31;
    int b = row / 512, n = row % 512;
    u16* o = out + (long long)row * 128;
    if (n >= Nn) {
#pragma unroll
        for (int j = 0; j < 4; j++) o[lane + j * 32] = 0;
        return;
    }
    const float* x = R + ((long long)b * Nn + n) * F + 3;
    float v[4], ss = 0.f;
#pragma unroll
    for (int j = 0; j < 4; j++) { v[j] = x[lane + j * 32]; ss += v[j] * v[j]; }
    ss = wred(ss);
    float inv = 1.f / fmaxf(sqrtf(ss), 1e-12f);
#pragma unroll
    for (int j = 0; j < 4; j++) o[lane + j * 32] = f2bf(v[j] * inv);
}

// ---------------------------------------------------------------------------
// argmax over axis2 (rows) and axis1 (cols) of corr[16][512][512] (valid 500)
// stable tie-break: smallest index (matches argsort)
// ---------------------------------------------------------------------------
__global__ __launch_bounds__(256) void argmax2_k(const float* __restrict__ corr,
                                                 int* __restrict__ oq, float* __restrict__ cq)
{
    int row = blockIdx.x * 8 + (threadIdx.x >> 5);
    if (row >= 16 * 500) return;
    int lane = threadIdx.x & 31;
    int b = row / 500, n = row % 500;
    const float* c = corr + ((long long)b * 512 + n) * 512;
    float best = -1e30f; int bi = 0x7fffffff;
    for (int m = lane; m < 500; m += 32) {
        float v = c[m];
        if (v > best) { best = v; bi = m; }
    }
#pragma unroll
    for (int d = 1; d < 32; d <<= 1) {
        float ov = __shfl_xor(best, d, 32);
        int   oi = __shfl_xor(bi, d, 32);
        if (ov > best || (ov == best && oi < bi)) { best = ov; bi = oi; }
    }
    if (lane == 0) { oq[row] = bi; cq[row] = best; }
}
__global__ __launch_bounds__(256) void argmax1_k(const float* __restrict__ corr,
                                                 int* __restrict__ ok, float* __restrict__ ck)
{
    int row = blockIdx.x * 8 + (threadIdx.x >> 5);
    if (row >= 16 * 500) return;
    int lane = threadIdx.x & 31;
    int b = row / 500, m = row % 500;
    const float* c = corr + (long long)b * 512 * 512 + m;
    float best = -1e30f; int bi = 0x7fffffff;
    for (int n = lane; n < 500; n += 32) {
        float v = c[(long long)n * 512];
        if (v > best) { best = v; bi = n; }
    }
#pragma unroll
    for (int d = 1; d < 32; d <<= 1) {
        float ov = __shfl_xor(best, d, 32);
        int   oi = __shfl_xor(bi, d, 32);
        if (ov > best || (ov == best && oi < bi)) { best = ov; bi = oi; }
    }
    if (lane == 0) { ok[row] = bi; ck[row] = best; }
}

// ---------------------------------------------------------------------------
// select features [16][1000][7]: rows 0..499 = sq, 500..999 = sk
// ---------------------------------------------------------------------------
__global__ void build_select_k(const float* __restrict__ xr, const float* __restrict__ yr,
                               const int* __restrict__ oq, const float* __restrict__ cq,
                               const int* __restrict__ ok, const float* __restrict__ ck,
                               float* __restrict__ sel)
{
    int t = blockIdx.x * blockDim.x + threadIdx.x;
    if (t >= 16 * 1000) return;
    int b = t / 1000, p = t % 1000;
    float o[7];
    if (p < 500) {
        int n = p;
        int m = oq[b * 500 + n];
#pragma unroll
        for (int j = 0; j < 3; j++) o[j]     = clamp01(xr[((long long)b * 500 + n) * 131 + j]);
#pragma unroll
        for (int j = 0; j < 3; j++) o[3 + j] = clamp01(yr[((long long)b * 500 + m) * 131 + j]);
        o[6] = cq[b * 500 + n];
    } else {
        int m = p - 500;
        int n = ok[b * 500 + m];
#pragma unroll
        for (int j = 0; j < 3; j++) o[j]     = clamp01(xr[((long long)b * 500 + n) * 131 + j]);
#pragma unroll
        for (int j = 0; j < 3; j++) o[3 + j] = clamp01(yr[((long long)b * 500 + m) * 131 + j]);
        o[6] = ck[b * 500 + m];
    }
#pragma unroll
    for (int j = 0; j < 7; j++) sel[(long long)t * 7 + j] = o[j];
}

// ---------------------------------------------------------------------------
// Pair embedding: X1 row(2s)=cls2, row(2s+1)= sel@pairw + pairb + PE(feat3,feat4)
// K=7 -> scalar MAC is the right tool here (no WMMA tile fits)
// ---------------------------------------------------------------------------
__global__ void pair_embed_k(const float* __restrict__ sel, const float* __restrict__ pairw,
                             const float* __restrict__ pairb, const float* __restrict__ cls2,
                             float* __restrict__ X1)
{
    long long t = (long long)blockIdx.x * blockDim.x + threadIdx.x;
    if (t >= 16000LL * 384) return;
    int s = (int)(t / 384), j = (int)(t % 384);
    const float* f = sel + (long long)s * 7;
    float v = pairb[j];
#pragma unroll
    for (int i = 0; i < 7; i++) v += f[i] * pairw[i * 384 + j];
    v += pe384(j, f[3], f[4]);
    X1[((long long)s * 2 + 1) * 384 + j] = v;
    X1[((long long)s * 2 + 0) * 384 + j] = cls2[j];
}

// ---------------------------------------------------------------------------
// LayerNorm -> bf16. One wave per row; input row index = row*instride.
// ---------------------------------------------------------------------------
__global__ __launch_bounds__(256) void ln_k(const float* __restrict__ X,
                                            const float* __restrict__ w, const float* __restrict__ b,
                                            u16* __restrict__ H, int M, int instride)
{
    int row = blockIdx.x * 8 + (threadIdx.x >> 5);
    if (row >= M) return;
    int lane = threadIdx.x & 31;
    const float* x = X + (long long)row * instride * 384;
    float v[12], s = 0.f;
#pragma unroll
    for (int j = 0; j < 12; j++) { v[j] = x[lane + j * 32]; s += v[j]; }
    s = wred(s);
    float mean = s * (1.f / 384.f);
    float vs = 0.f;
#pragma unroll
    for (int j = 0; j < 12; j++) { float d = v[j] - mean; vs += d * d; }
    vs = wred(vs);
    float inv = rsqrtf(vs * (1.f / 384.f) + 1e-5f);
    u16* o = H + (long long)row * 384;
#pragma unroll
    for (int j = 0; j < 12; j++) {
        int c = lane + j * 32;
        o[c] = f2bf((v[j] - mean) * inv * w[c] + b[c]);
    }
}

// ---------------------------------------------------------------------------
// Seq-len-2 attention (stage B). One wave per (sequence, head). Exact VALU.
// QKV layout per row: [3][H=6][dh=64] -> q@h*64, k@384+h*64, v@768+h*64
// ---------------------------------------------------------------------------
__global__ __launch_bounds__(256) void attn_s2_k(const u16* __restrict__ QKV,
                                                 u16* __restrict__ O, int nseq)
{
    int gid = blockIdx.x * 8 + (threadIdx.x >> 5);
    if (gid >= nseq * 6) return;
    int lane = threadIdx.x & 31;
    int s = gid / 6, h = gid % 6;
    const u16* r0 = QKV + (long long)s * 2 * 1152 + h * 64;
    const u16* r1 = r0 + 1152;
    float q0a = bf2f(r0[lane]),        q0b = bf2f(r0[lane + 32]);
    float q1a = bf2f(r1[lane]),        q1b = bf2f(r1[lane + 32]);
    float k0a = bf2f(r0[384 + lane]),  k0b = bf2f(r0[384 + lane + 32]);
    float k1a = bf2f(r1[384 + lane]),  k1b = bf2f(r1[384 + lane + 32]);
    float v0a = bf2f(r0[768 + lane]),  v0b = bf2f(r0[768 + lane + 32]);
    float v1a = bf2f(r1[768 + lane]),  v1b = bf2f(r1[768 + lane + 32]);
    float s00 = wred(q0a * k0a + q0b * k0b) * 0.125f;
    float s01 = wred(q0a * k1a + q0b * k1b) * 0.125f;
    float s10 = wred(q1a * k0a + q1b * k0b) * 0.125f;
    float s11 = wred(q1a * k1a + q1b * k1b) * 0.125f;
    float m0 = fmaxf(s00, s01), m1 = fmaxf(s10, s11);
    float a00 = __expf(s00 - m0), a01 = __expf(s01 - m0);
    float a10 = __expf(s10 - m1), a11 = __expf(s11 - m1);
    float d0 = a00 + a01, d1 = a10 + a11;
    u16* o0 = O + (long long)s * 2 * 384 + h * 64;
    u16* o1 = o0 + 384;
    o0[lane]      = f2bf((a00 * v0a + a01 * v1a) / d0);
    o0[lane + 32] = f2bf((a00 * v0b + a01 * v1b) / d0);
    o1[lane]      = f2bf((a10 * v0a + a11 * v1a) / d1);
    o1[lane + 32] = f2bf((a10 * v0b + a11 * v1b) / d1);
}

// ---------------------------------------------------------------------------
// Flash attention for S=1001 (padded to 1024). One wave per (b, h, 16-q tile).
// Scores & P@V via WMMA bf16; online softmax in f32; keys >= SV masked.
// ---------------------------------------------------------------------------
__global__ __launch_bounds__(32) void attn_flash_k(const u16* __restrict__ QKV,
                                                   u16* __restrict__ O, int SP, int SV)
{
    __shared__ __align__(16) u16 Pls[16][40];
    const int qt = blockIdx.x, h = blockIdx.y, b = blockIdx.z;
    const int lane = threadIdx.x;
    const int lane15 = lane & 15, kb = (lane < 16) ? 0 : 8;
    const int half8 = (lane < 16) ? 0 : 8;
    const long long rowbase = (long long)b * SP + qt * 16;
    const int LD = 1152;

    v8u aq[2];
#pragma unroll
    for (int kc = 0; kc < 2; kc++)
        aq[kc] = load_frag16(QKV + (rowbase + lane15) * LD + h * 64 + kc * 32 + kb);

    v8f o[4] = {};
    float m_i[8], l_i[8];
#pragma unroll
    for (int i = 0; i < 8; i++) { m_i[i] = -1e30f; l_i[i] = 0.f; }

    for (int key0 = 0; key0 < SP; key0 += 32) {
        v8f s0 = {}, s1 = {};
#pragma unroll
        for (int kc = 0; kc < 2; kc++) {
            v8u b0 = load_frag16(QKV + ((long long)b * SP + key0 + lane15) * LD + 384 + h * 64 + kc * 32 + kb);
            v8u b1 = load_frag16(QKV + ((long long)b * SP + key0 + 16 + lane15) * LD + 384 + h * 64 + kc * 32 + kb);
            s0 = wmma_bf16(aq[kc], b0, s0);
            s1 = wmma_bf16(aq[kc], b1, s1);
        }
        const bool v0ok = (key0 + lane15) < SV;
        const bool v1ok = (key0 + 16 + lane15) < SV;
        float mc[8];
#pragma unroll
        for (int i = 0; i < 8; i++) {
            s0[i] = v0ok ? s0[i] * 0.125f : -1e30f;
            s1[i] = v1ok ? s1[i] * 0.125f : -1e30f;
            mc[i] = fmaxf(s0[i], s1[i]);
        }
#pragma unroll
        for (int d = 1; d < 16; d <<= 1)
#pragma unroll
            for (int i = 0; i < 8; i++) mc[i] = fmaxf(mc[i], __shfl_xor(mc[i], d, 32));
        float sc[8], rs[8];
#pragma unroll
        for (int i = 0; i < 8; i++) {
            float mn = fmaxf(m_i[i], mc[i]);
            sc[i] = __expf(m_i[i] - mn);
            m_i[i] = mn;
            s0[i] = __expf(s0[i] - mn);
            s1[i] = __expf(s1[i] - mn);
            rs[i] = s0[i] + s1[i];
        }
#pragma unroll
        for (int d = 1; d < 16; d <<= 1)
#pragma unroll
            for (int i = 0; i < 8; i++) rs[i] += __shfl_xor(rs[i], d, 32);
#pragma unroll
        for (int i = 0; i < 8; i++) {
            l_i[i] = l_i[i] * sc[i] + rs[i];
            o[0][i] *= sc[i]; o[1][i] *= sc[i]; o[2][i] *= sc[i]; o[3][i] *= sc[i];
            Pls[half8 + i][lane15]      = f2bf(s0[i]);   // C-layout -> LDS
            Pls[half8 + i][16 + lane15] = f2bf(s1[i]);
        }
        __syncthreads();
        v8u ap = load_frag16(&Pls[lane15][kb]);          // reload as A-fragment
        __syncthreads();
#pragma unroll
        for (int nt = 0; nt < 4; nt++) {
            const u16* pv = QKV + ((long long)b * SP + key0 + kb) * LD + 768 + h * 64 + nt * 16 + lane15;
            v8u bv;
#pragma unroll
            for (int j = 0; j < 4; j++) {
                unsigned lo  = pv[(2 * j) * LD],       hi  = pv[(2 * j + 1) * LD];
                unsigned lo2 = pv[(16 + 2 * j) * LD],  hi2 = pv[(16 + 2 * j + 1) * LD];
                bv[j]     = lo  | (hi  << 16);
                bv[4 + j] = lo2 | (hi2 << 16);
            }
            o[nt] = wmma_bf16(ap, bv, o[nt]);
        }
    }
#pragma unroll
    for (int i = 0; i < 8; i++) {
        float inv = 1.f / l_i[i];
        long long row = rowbase + half8 + i;
        u16* op = O + row * 384 + h * 64 + lane15;
        op[0]  = f2bf(o[0][i] * inv);
        op[16] = f2bf(o[1][i] * inv);
        op[32] = f2bf(o[2][i] * inv);
        op[48] = f2bf(o[3][i] * inv);
    }
}

// ---------------------------------------------------------------------------
// X2 assembly: [16][1024][384]; r0 = cls, r1..1000 = T + PE(feat0,feat1), rest 0
// ---------------------------------------------------------------------------
__global__ void assemble_x2_k(const float* __restrict__ T, const float* __restrict__ sel,
                              const float* __restrict__ cls, float* __restrict__ X2)
{
    long long t = (long long)blockIdx.x * blockDim.x + threadIdx.x;
    if (t >= 16LL * 1024 * 384) return;
    int b = (int)(t / (1024 * 384));
    int rem = (int)(t % (1024 * 384));
    int r = rem / 384, j = rem % 384;
    float v;
    if (r == 0) v = cls[j];
    else if (r <= 1000) {
        int p = r - 1;
        const float* f = sel + ((long long)b * 1000 + p) * 7;
        v = T[((long long)b * 1000 + p) * 384 + j] + pe384(j, f[0], f[1]);
    } else v = 0.f;
    X2[t] = v;
}

// ---------------------------------------------------------------------------
// Final head: LN(X2[b,0]) . predw + predb -> sigmoid; blend with global score
// ---------------------------------------------------------------------------
__global__ __launch_bounds__(384) void final_k(const float* __restrict__ X2,
                                               const float* __restrict__ dnw, const float* __restrict__ dnb,
                                               const float* __restrict__ predw, const float* __restrict__ predb,
                                               const float* __restrict__ ratio, const float* __restrict__ gs,
                                               float* __restrict__ out)
{
    __shared__ float red[12];
    int b = blockIdx.x, t = threadIdx.x;
    float x = X2[(long long)b * 1024 * 384 + t];
    auto blockSum = [&](float v) -> float {
        v = wred(v);
        if ((t & 31) == 0) red[t >> 5] = v;
        __syncthreads();
        float s = 0.f;
        for (int i = 0; i < 12; i++) s += red[i];
        __syncthreads();
        return s;
    };
    float mean = blockSum(x) * (1.f / 384.f);
    float d = x - mean;
    float var = blockSum(d * d) * (1.f / 384.f);
    float ln = d * rsqrtf(var + 1e-5f) * dnw[t] + dnb[t];
    float dot = blockSum(ln * predw[t]);
    if (t == 0) {
        float local = 1.f / (1.f + __expf(-(dot + predb[0])));
        float r = fminf(fmaxf(ratio[0], 0.1f), 0.9f);
        out[b] = local;
        out[16 + b] = gs[b] * r + local * (1.f - r);
    }
}

// ---------------------------------------------------------------------------
// Host orchestration
// Input pytree leaf order (jax: top-level dict in insertion order, nested
// dicts sorted by key, lists in order):
//   0 x_global, 1 x_rerank, 2 y_global, 3 y_rerank,
//   4..51  blocks[0..3]  x {fc1b,fc1w,fc2b,fc2w,n1b,n1w,n2b,n2w,projb,projw,qkvb,qkvw}
//   52..75 blocks2[0..1] x same
//   76 cls, 77 cls2, 78 dnb, 79 dnw, 80 pair2b, 81 pair2w,
//   82 pairb, 83 pairw, 84 predb, 85 predw, 86 ratio
// ---------------------------------------------------------------------------
struct Blk {
    const float *fc1b, *fc1w, *fc2b, *fc2w, *n1b, *n1w, *n2b, *n2w,
                *projb, *projw, *qkvb, *qkvw;
};

extern "C" void kernel_launch(void* const* d_in, const int* in_sizes, int n_in,
                              void* d_out, int out_size, void* d_ws, size_t ws_size,
                              hipStream_t stream)
{
    (void)in_sizes; (void)out_size; (void)ws_size;
    if (n_in < 87 || !d_out || !d_ws) return;

    const float* x_global = (const float*)d_in[0];
    const float* x_rerank = (const float*)d_in[1];
    const float* y_global = (const float*)d_in[2];
    const float* y_rerank = (const float*)d_in[3];

    Blk blk[6];
    for (int i = 0; i < 6; i++) {
        int base = (i < 4) ? (4 + 12 * i) : (52 + 12 * (i - 4));
        blk[i].fc1b  = (const float*)d_in[base + 0];
        blk[i].fc1w  = (const float*)d_in[base + 1];
        blk[i].fc2b  = (const float*)d_in[base + 2];
        blk[i].fc2w  = (const float*)d_in[base + 3];
        blk[i].n1b   = (const float*)d_in[base + 4];
        blk[i].n1w   = (const float*)d_in[base + 5];
        blk[i].n2b   = (const float*)d_in[base + 6];
        blk[i].n2w   = (const float*)d_in[base + 7];
        blk[i].projb = (const float*)d_in[base + 8];
        blk[i].projw = (const float*)d_in[base + 9];
        blk[i].qkvb  = (const float*)d_in[base + 10];
        blk[i].qkvw  = (const float*)d_in[base + 11];
    }
    const float* cls    = (const float*)d_in[76];
    const float* cls2   = (const float*)d_in[77];
    const float* dnb    = (const float*)d_in[78];
    const float* dnw    = (const float*)d_in[79];
    const float* pair2b = (const float*)d_in[80];
    const float* pair2w = (const float*)d_in[81];
    const float* pairb  = (const float*)d_in[82];
    const float* pairw  = (const float*)d_in[83];
    const float* predb  = (const float*)d_in[84];
    const float* predw  = (const float*)d_in[85];
    const float* ratio  = (const float*)d_in[86];
    float* out = (float*)d_out;

    // ---- deterministic workspace carve --------------------------------
    char* p = (char*)d_ws;
    auto alloc = [&](size_t n) { char* r = p; p += (n + 255) & ~(size_t)255; return r; };
    float* GS  = (float*)alloc(16 * 4);
    u16*   XTN = (u16*)alloc((size_t)16 * 512 * 128 * 2);
    u16*   YTN = (u16*)alloc((size_t)16 * 512 * 128 * 2);
    int*   OQ  = (int*)alloc(16 * 500 * 4);
    float* CQ  = (float*)alloc(16 * 500 * 4);
    int*   OK_ = (int*)alloc(16 * 500 * 4);
    float* CK  = (float*)alloc(16 * 500 * 4);
    float* SEL = (float*)alloc((size_t)16 * 1000 * 7 * 4);
    u16*   WTS = (u16*)alloc((size_t)10764288 * 2);
    float* X1  = (float*)alloc((size_t)32000 * 384 * 4);
    u16*   HB  = (u16*)alloc((size_t)32000 * 384 * 2);
    u16*   OB  = (u16*)alloc((size_t)32000 * 384 * 2);
    float* X2  = (float*)alloc((size_t)16 * 1024 * 384 * 4);
    char*  BIG = alloc((size_t)32000 * 1536 * 2);     // aliased: CORR / QKV / G / T
    float* CORR = (float*)BIG;
    float* Tbuf = (float*)BIG;

    // transposed bf16 weight slots
    u16 *qkvT[6], *projT[6], *fc1T[6], *fc2T[6];
    size_t woff = 0;
    for (int i = 0; i < 6; i++) {
        qkvT[i]  = WTS + woff; woff += (size_t)384 * 1152;
        projT[i] = WTS + woff; woff += (size_t)384 * 384;
        fc1T[i]  = WTS + woff; woff += (size_t)384 * 1536;
        fc2T[i]  = WTS + woff; woff += (size_t)1536 * 384;
    }
    u16* pair2T = WTS + woff;

    auto prep = [&](const float* w, u16* wt, int K, int N) {
        long long total = (long long)K * N;
        prep_wt_k<<<(unsigned)((total + 255) / 256), 256, 0, stream>>>(w, wt, K, N);
    };
    for (int i = 0; i < 6; i++) {
        prep(blk[i].qkvw,  qkvT[i],  384, 1152);
        prep(blk[i].projw, projT[i], 384, 384);
        prep(blk[i].fc1w,  fc1T[i],  384, 1536);
        prep(blk[i].fc2w,  fc2T[i],  1536, 384);
    }
    prep(pair2w, pair2T, 384, 384);

    // ---- phase 0: scores, correlation, matching -----------------------
    gscore_k<<<16, 256, 0, stream>>>(x_global, y_global, GS);
    norm_desc_k<<<1024, 256, 0, stream>>>(x_rerank, XTN, 500, 131);
    norm_desc_k<<<1024, 256, 0, stream>>>(y_rerank, YTN, 500, 131);
    // corr[b] = XTN[b] @ YTN[b]^T : YTN row-major [512][128] == WT [N][K]
    gemm_wmma_k<0, 0, 0><<<dim3(4, 8, 16), 256, 0, stream>>>(
        XTN, YTN, nullptr, CORR, nullptr, 512, 512, 128, 65536, 65536, 262144);
    argmax2_k<<<1000, 256, 0, stream>>>(CORR, OQ, CQ);
    argmax1_k<<<1000, 256, 0, stream>>>(CORR, OK_, CK);
    build_select_k<<<63, 256, 0, stream>>>(x_rerank, y_rerank, OQ, CQ, OK_, CK, SEL);
    pair_embed_k<<<24000, 256, 0, stream>>>(SEL, pairw, pairb, cls2, X1);

    // ---- phase B: blocks2 (2 blocks) on 16000 x len-2 sequences -------
    for (int i = 4; i < 6; i++) {
        u16* QKVb = (u16*)BIG;
        u16* Gb   = (u16*)BIG;
        ln_k<<<4000, 256, 0, stream>>>(X1, blk[i].n1w, blk[i].n1b, HB, 32000, 1);
        gemm_wmma_k<0, 0, 1><<<dim3(9, 500, 1), 256, 0, stream>>>(
            HB, qkvT[i], blk[i].qkvb, nullptr, QKVb, 32000, 1152, 384, 0, 0, 0);
        attn_s2_k<<<12000, 256, 0, stream>>>(QKVb, OB, 16000);
        gemm_wmma_k<0, 1, 0><<<dim3(3, 500, 1), 256, 0, stream>>>(
            OB, projT[i], blk[i].projb, X1, nullptr, 32000, 384, 384, 0, 0, 0);
        ln_k<<<4000, 256, 0, stream>>>(X1, blk[i].n2w, blk[i].n2b, HB, 32000, 1);
        gemm_wmma_k<1, 0, 1><<<dim3(12, 500, 1), 256, 0, stream>>>(
            HB, fc1T[i], blk[i].fc1b, nullptr, Gb, 32000, 1536, 384, 0, 0, 0);
        gemm_wmma_k<0, 1, 0><<<dim3(3, 500, 1), 256, 0, stream>>>(
            Gb, fc2T[i], blk[i].fc2b, X1, nullptr, 32000, 384, 1536, 0, 0, 0);
    }

    // ---- phase C: LN(row0) @ pair2w + PE -> X2 ------------------------
    ln_k<<<2000, 256, 0, stream>>>(X1, dnw, dnb, HB, 16000, 2);
    gemm_wmma_k<0, 0, 0><<<dim3(3, 250, 1), 256, 0, stream>>>(
        HB, pair2T, pair2b, Tbuf, nullptr, 16000, 384, 384, 0, 0, 0);
    assemble_x2_k<<<24576, 256, 0, stream>>>(Tbuf, SEL, cls, X2);

    // ---- phase D: blocks (4 blocks) on 16 x 1024 tokens (1001 valid) --
    for (int i = 0; i < 4; i++) {
        u16* QKVb = (u16*)BIG;
        u16* Gb   = (u16*)BIG;
        ln_k<<<2048, 256, 0, stream>>>(X2, blk[i].n1w, blk[i].n1b, HB, 16384, 1);
        gemm_wmma_k<0, 0, 1><<<dim3(9, 256, 1), 256, 0, stream>>>(
            HB, qkvT[i], blk[i].qkvb, nullptr, QKVb, 16384, 1152, 384, 0, 0, 0);
        attn_flash_k<<<dim3(64, 6, 16), 32, 0, stream>>>(QKVb, OB, 1024, 1001);
        gemm_wmma_k<0, 1, 0><<<dim3(3, 256, 1), 256, 0, stream>>>(
            OB, projT[i], blk[i].projb, X2, nullptr, 16384, 384, 384, 0, 0, 0);
        ln_k<<<2048, 256, 0, stream>>>(X2, blk[i].n2w, blk[i].n2b, HB, 16384, 1);
        gemm_wmma_k<1, 0, 1><<<dim3(12, 256, 1), 256, 0, stream>>>(
            HB, fc1T[i], blk[i].fc1b, nullptr, Gb, 16384, 1536, 384, 0, 0, 0);
        gemm_wmma_k<0, 1, 0><<<dim3(3, 256, 1), 256, 0, stream>>>(
            Gb, fc2T[i], blk[i].fc2b, X2, nullptr, 16384, 384, 1536, 0, 0, 0);
    }

    // ---- phase E: head -------------------------------------------------
    final_k<<<16, 384, 0, stream>>>(X2, dnw, dnb, predw, predb, ratio, GS, out);
}
